// LeftRightGNN_7902739825340
// MI455X (gfx1250) — matrix operand order repaired and checked by reading, not compiled, not run on hardware
//
#include <hip/hip_runtime.h>
#include <hip/hip_bf16.h>
#include <math.h>

// ---------------------------------------------------------------------------
// MI455X (gfx1250) implementation.
// Precision plan: all GEMM-like work via v_wmma_f32_16x16x32_bf16 (bf16 in,
// f32 accumulate). Scatter/softmax/segment ops stay f32 VALU + f32 atomics.
// Transcendentals are branch-free (v_exp_f32 + v_rcp_f32) to keep the LSTM
// elementwise tail off the critical path between WMMA bursts.
// ---------------------------------------------------------------------------

typedef __attribute__((ext_vector_type(16))) __bf16          v16bf;
typedef __attribute__((ext_vector_type(16))) unsigned short  v16us;
typedef __attribute__((ext_vector_type(8)))  float           v8f;

union Frag { v16us u; v16bf b; };

__device__ __forceinline__ unsigned short f2bf(float f) {
  unsigned int x = __float_as_uint(f);
  unsigned int r = x + 0x7fffu + ((x >> 16) & 1u);   // round-to-nearest-even
  return (unsigned short)(r >> 16);
}
__device__ __forceinline__ float bf2f(unsigned short u) {
  return __uint_as_float(((unsigned int)u) << 16);
}
// A-matrix (16-bit, 16x32) element k for (vgpr vi, half h, lane-group hi)
__device__ __forceinline__ int ka_idx(int vi, int h, int hi) {
  return ((vi >> 2) << 4) + (hi << 3) + ((vi & 3) << 1) + h;
}
// B-matrix (16-bit, 32x16) element k
__device__ __forceinline__ int kb_idx(int vi, int h, int hi) {
  return ((vi >> 2) << 4) + (hi << 3) + ((vi & 3) << 1) + h;
}
// Branch-free transcendentals: v_exp_f32 + v_rcp_f32, no exec-mask churn.
__device__ __forceinline__ float sigm(float x) {
  return __builtin_amdgcn_rcpf(1.0f + __expf(-x));     // exp->inf => 0, correct
}
__device__ __forceinline__ float tanh_fast(float x) {
  return 1.0f - 2.0f * __builtin_amdgcn_rcpf(__expf(2.0f * x) + 1.0f);
}

__device__ __forceinline__ void atomicMaxFloat(float* addr, float val) {
  if (val >= 0.0f) atomicMax((int*)addr, __float_as_int(val));
  else             atomicMin((unsigned int*)addr, __float_as_uint(val));
}

// ------------------------- constants ---------------------------------------
#define NN   20000
#define EE   100000
#define TT   32
#define DEMB 32
#define HL   64
#define SDD  53
#define ST   181   // SD + 128
#define IN1  417
#define IN1P 448
#define CHN  256
#define EDIM 128

// ------------------------- weight pack (f32 -> bf16 [Kpad][N]) -------------
__global__ void pack_b_kernel(const float* __restrict__ src,
                              unsigned short* __restrict__ dst,
                              int K, int Nc, int Kpad, int trans) {
  int tid = blockIdx.x * blockDim.x + threadIdx.x;
  if (tid >= Kpad * Nc) return;
  int k = tid / Nc, n = tid % Nc;
  float v = 0.0f;
  if (k < K) v = trans ? src[n * K + k] : src[k * Nc + n];
  dst[tid] = f2bf(v);
}

// ------------------------- LSTM (fwd scan + 1-step bwd) --------------------
// One 16-edge tile per block; 4 waves, wave j owns hidden cols [16j,16j+16).
// Weights register-resident as WMMA B fragments; h round-trips via LDS bf16.
__global__ __launch_bounds__(128) void lstm_kernel(
    const int* __restrict__ tokens, const float* __restrict__ embed,
    const unsigned short* __restrict__ BihF, const unsigned short* __restrict__ BhhF,
    const float* __restrict__ bF,
    const unsigned short* __restrict__ BihB, const float* __restrict__ bB,
    float* __restrict__ regex, int E_) {
  __shared__ unsigned short embL[30 * 32];
  __shared__ unsigned short hL[16 * 64];
  const int tid  = threadIdx.x;
  const int wave = tid >> 5;
  const int lane = tid & 31;
  const int col  = lane & 15;
  const int hi   = lane >> 4;
  const int ebase = blockIdx.x * 16;

  for (int i = tid; i < 30 * 32; i += 128) embL[i] = f2bf(embed[i]);
  for (int i = tid; i < 16 * 64; i += 128) hL[i] = 0;
  __syncthreads();

  // Register-resident B fragments for this wave's 4 gate tiles.
  Frag wih[4], whh0[4], whh1[4];
  float bias[4];
#pragma unroll
  for (int g = 0; g < 4; ++g) {
    const int nb = g * 64 + wave * 16;
    bias[g] = bF[nb + col];
#pragma unroll
    for (int vi = 0; vi < 8; ++vi)
#pragma unroll
      for (int h2 = 0; h2 < 2; ++h2) {
        const int kk = kb_idx(vi, h2, hi);
        wih[g].u[2 * vi + h2]  = BihF[kk * 256 + nb + col];
        whh0[g].u[2 * vi + h2] = BhhF[kk * 256 + nb + col];
        whh1[g].u[2 * vi + h2] = BhhF[(kk + 32) * 256 + nb + col];
      }
  }

  const int m = col;                 // A-matrix row owned by this lane
  const int edge = ebase + m;
  v8f cst;
#pragma unroll
  for (int r = 0; r < 8; ++r) cst[r] = 0.0f;

  for (int t = 0; t < TT; ++t) {
    Frag xa, ha0, ha1;
    const int tok = (edge < E_) ? tokens[edge * TT + t] : 0;
#pragma unroll
    for (int vi = 0; vi < 8; ++vi)
#pragma unroll
      for (int h2 = 0; h2 < 2; ++h2) {
        const int kk = ka_idx(vi, h2, hi);
        xa.u[2 * vi + h2]  = embL[tok * 32 + kk];
        ha0.u[2 * vi + h2] = hL[m * 64 + kk];
        ha1.u[2 * vi + h2] = hL[m * 64 + 32 + kk];
      }
    __syncthreads();   // all waves done reading h

    v8f acc[4];
#pragma unroll
    for (int g = 0; g < 4; ++g) {
#pragma unroll
      for (int r = 0; r < 8; ++r) acc[g][r] = bias[g];
      acc[g] = __builtin_amdgcn_wmma_f32_16x16x32_bf16(false, xa.b,  false, wih[g].b,  (short)0, acc[g], false, false);
      acc[g] = __builtin_amdgcn_wmma_f32_16x16x32_bf16(false, ha0.b, false, whh0[g].b, (short)0, acc[g], false, false);
      acc[g] = __builtin_amdgcn_wmma_f32_16x16x32_bf16(false, ha1.b, false, whh1[g].b, (short)0, acc[g], false, false);
    }
#pragma unroll
    for (int r = 0; r < 8; ++r) {
      const float iv = sigm(acc[0][r]);
      const float fv = sigm(acc[1][r]);
      const float gv = tanh_fast(acc[2][r]);
      const float ov = sigm(acc[3][r]);
      cst[r] = fv * cst[r] + iv * gv;
      const float hn = ov * tanh_fast(cst[r]);
      hL[(r + 8 * hi) * 64 + wave * 16 + col] = f2bf(hn);
    }
    __syncthreads();   // h published for next step
  }

  // regex[:, 0:64] = final forward hidden
  for (int i = tid; i < 16 * 64; i += 128) {
    const int r = i >> 6, cc = i & 63;
    const int e2 = ebase + r;
    if (e2 < E_) regex[e2 * EDIM + cc] = bf2f(hL[i]);
  }

  // Backward direction: single cell on emb[:, T-1] with h=c=0 (W_hh unused).
  Frag wb[4];
  float bb2[4];
#pragma unroll
  for (int g = 0; g < 4; ++g) {
    const int nb = g * 64 + wave * 16;
    bb2[g] = bB[nb + col];
#pragma unroll
    for (int vi = 0; vi < 8; ++vi)
#pragma unroll
      for (int h2 = 0; h2 < 2; ++h2)
        wb[g].u[2 * vi + h2] = BihB[kb_idx(vi, h2, hi) * 256 + nb + col];
  }
  Frag xb;
  const int tokl = (edge < E_) ? tokens[edge * TT + (TT - 1)] : 0;
#pragma unroll
  for (int vi = 0; vi < 8; ++vi)
#pragma unroll
    for (int h2 = 0; h2 < 2; ++h2)
      xb.u[2 * vi + h2] = embL[tokl * 32 + ka_idx(vi, h2, hi)];

  v8f ab[4];
#pragma unroll
  for (int g = 0; g < 4; ++g) {
#pragma unroll
    for (int r = 0; r < 8; ++r) ab[g][r] = bb2[g];
    ab[g] = __builtin_amdgcn_wmma_f32_16x16x32_bf16(false, xb.b, false, wb[g].b, (short)0, ab[g], false, false);
  }
#pragma unroll
  for (int r = 0; r < 8; ++r) {
    const float c2 = sigm(ab[0][r]) * tanh_fast(ab[2][r]);
    const float hb = sigm(ab[3][r]) * tanh_fast(c2);
    const int e2 = ebase + r + 8 * hi;
    if (e2 < E_) regex[e2 * EDIM + 64 + wave * 16 + col] = hb;
  }
}

// ------------------------- generic bf16 WMMA GEMM --------------------------
// C[M,256] = A[M,K] @ B[Kpad,256] (+bias). 4 waves/block, wave w owns
// n-tile (blockIdx.y*4+w); A tile staged f32->bf16 through LDS.
__global__ __launch_bounds__(128) void gemm_kernel(
    const float* __restrict__ A, const unsigned short* __restrict__ B,
    const float* __restrict__ bias, float* __restrict__ C,
    int M, int K, int Kpad, int Nc) {
  __shared__ unsigned short aL[16 * 32];
  const int tid = threadIdx.x, wave = tid >> 5, lane = tid & 31;
  const int col = lane & 15, hi = lane >> 4;
  const int mbase = blockIdx.x * 16;
  const int nb = (blockIdx.y * 4 + wave) * 16;

  v8f acc;
  const float binit = bias ? bias[nb + col] : 0.0f;
#pragma unroll
  for (int r = 0; r < 8; ++r) acc[r] = binit;

  for (int k0 = 0; k0 < Kpad; k0 += 32) {
    __syncthreads();
    for (int i = tid; i < 512; i += 128) {
      const int r = i >> 5, kk = k0 + (i & 31);
      const int mm = mbase + r;
      aL[i] = f2bf((kk < K && mm < M) ? A[(long)mm * K + kk] : 0.0f);
    }
    __syncthreads();
    Frag af, bfr;
#pragma unroll
    for (int vi = 0; vi < 8; ++vi)
#pragma unroll
      for (int h2 = 0; h2 < 2; ++h2) {
        af.u[2 * vi + h2]  = aL[col * 32 + ka_idx(vi, h2, hi)];
        bfr.u[2 * vi + h2] = B[(long)(k0 + kb_idx(vi, h2, hi)) * Nc + nb + col];
      }
    acc = __builtin_amdgcn_wmma_f32_16x16x32_bf16(false, af.b, false, bfr.b, (short)0, acc, false, false);
  }
#pragma unroll
  for (int r = 0; r < 8; ++r) {
    const int mm = mbase + r + 8 * hi;
    if (mm < M) C[(long)mm * Nc + nb + col] = acc[r];
  }
}

// ------------------------- segment means -----------------------------------
__global__ void seg_scatter_kernel(const float* __restrict__ src_state,
                                   const float* __restrict__ tgt_state,
                                   const float* __restrict__ regex,
                                   const int* __restrict__ ei,
                                   float* __restrict__ in_tr, float* __restrict__ out_tr,
                                   float* __restrict__ cnt_src, float* __restrict__ cnt_dst,
                                   int E_) {
  long tid = (long)blockIdx.x * blockDim.x + threadIdx.x;
  const long half = (long)E_ * ST;
  if (tid >= 2 * half) return;
  const int which = tid >= half;
  const long t2 = which ? tid - half : tid;
  const int e = (int)(t2 / ST), f = (int)(t2 % ST);
  if (!which) {           // out_tr = mean_{src}( [tgt_state, regex] )
    const int s = ei[e];
    const float v = (f < SDD) ? tgt_state[e * SDD + f] : regex[e * EDIM + (f - SDD)];
    atomicAdd(&out_tr[s * ST + f], v);
    if (f == 0) atomicAdd(&cnt_src[s], 1.0f);
  } else {                // in_tr = mean_{dst}( [src_state, regex] )
    const int d = ei[E_ + e];
    const float v = (f < SDD) ? src_state[e * SDD + f] : regex[e * EDIM + (f - SDD)];
    atomicAdd(&in_tr[d * ST + f], v);
    if (f == 0) atomicAdd(&cnt_dst[d], 1.0f);
  }
}

__global__ void seg_div_kernel(float* __restrict__ in_tr, float* __restrict__ out_tr,
                               const float* __restrict__ cnt_src,
                               const float* __restrict__ cnt_dst, int Nn) {
  int tid = blockIdx.x * blockDim.x + threadIdx.x;
  if (tid >= Nn * ST) return;
  const int n = tid / ST;
  out_tr[tid] *= __builtin_amdgcn_rcpf(fmaxf(cnt_src[n], 1.0f));
  in_tr[tid]  *= __builtin_amdgcn_rcpf(fmaxf(cnt_dst[n], 1.0f));
}

__global__ void concat_kernel(const float* __restrict__ left_x, const float* __restrict__ right_x,
                              const float* __restrict__ in_tr, const float* __restrict__ out_tr,
                              float* __restrict__ lx, float* __restrict__ rx, int Nn) {
  int tid = blockIdx.x * blockDim.x + threadIdx.x;
  if (tid >= Nn * IN1) return;
  const int n = tid / IN1, f = tid % IN1;
  float a, b;
  if (f < SDD + 2)      { a = left_x[n * (SDD + 2) + f];  b = right_x[n * (SDD + 2) + f]; }
  else if (f < SDD + 2 + ST) { const int g = f - (SDD + 2); a = in_tr[n * ST + g];  b = out_tr[n * ST + g]; }
  else                  { const int g = f - (SDD + 2 + ST); a = out_tr[n * ST + g]; b = in_tr[n * ST + g]; }
  lx[tid] = a; rx[tid] = b;
}

// ------------------------- GAT attention passes ----------------------------
__global__ void att_init_kernel(float* __restrict__ mx, float* __restrict__ den,
                                float* __restrict__ outacc, const float* __restrict__ bias,
                                int Nn) {
  int tid = blockIdx.x * blockDim.x + threadIdx.x;
  if (tid >= Nn * CHN) return;
  const int n = tid / CHN, cch = tid % CHN;
  outacc[tid] = bias[cch];
  if (cch < 8) { mx[n * 8 + cch] = -1e38f; den[n * 8 + cch] = 0.0f; }
}

__global__ void logits_kernel(const float* __restrict__ xl, const float* __restrict__ xr,
                              const float* __restrict__ ee, const float* __restrict__ att,
                              const int* __restrict__ ei,
                              float* __restrict__ logit, float* __restrict__ mx, int E_) {
  int tid = blockIdx.x * blockDim.x + threadIdx.x;
  if (tid >= E_ * 8) return;
  const int e = tid / 8, h = tid % 8;
  const int s = ei[e], d = ei[E_ + e];
  float acc = 0.0f;
#pragma unroll 8
  for (int cc = 0; cc < 32; ++cc) {
    float m = xl[s * CHN + h * 32 + cc] + xr[d * CHN + h * 32 + cc] + ee[(long)e * CHN + h * 32 + cc];
    m = (m > 0.0f) ? m : 0.2f * m;                  // leaky_relu(0.2)
    acc += m * att[h * 32 + cc];
  }
  logit[tid] = acc;
  atomicMaxFloat(&mx[d * 8 + h], acc);
}

__global__ void expden_kernel(const float* __restrict__ logit, const float* __restrict__ mx,
                              const int* __restrict__ ei,
                              float* __restrict__ exb, float* __restrict__ den, int E_) {
  int tid = blockIdx.x * blockDim.x + threadIdx.x;
  if (tid >= E_ * 8) return;
  const int e = tid / 8, h = tid % 8;
  const int d = ei[E_ + e];
  float m = mx[d * 8 + h];
  if (!(m > -1e37f)) m = 0.0f;                      // mirror where(isfinite(mx), mx, 0)
  const float ex = __expf(logit[tid] - m);
  exb[tid] = ex;
  atomicAdd(&den[d * 8 + h], ex);
}

__global__ void attout_kernel(const float* __restrict__ xl, const float* __restrict__ exb,
                              const float* __restrict__ den, const int* __restrict__ ei,
                              float* __restrict__ outacc, int E_) {
  long tid = (long)blockIdx.x * blockDim.x + threadIdx.x;
  if (tid >= (long)E_ * CHN) return;
  const int e = (int)(tid / CHN), cch = (int)(tid % CHN);
  const int h = cch >> 5;
  const int s = ei[e], d = ei[E_ + e];
  const float alpha = exb[e * 8 + h] * __builtin_amdgcn_rcpf(den[d * 8 + h] + 1e-16f);
  atomicAdd(&outacc[d * CHN + cch], alpha * xl[s * CHN + cch]);
}

__global__ void elu_kernel(const float* __restrict__ in, float* __restrict__ out, int total) {
  int tid = blockIdx.x * blockDim.x + threadIdx.x;
  if (tid >= total) return;
  const float x = in[tid];
  out[tid] = (x > 0.0f) ? x : (__expf(x) - 1.0f);
}

__global__ void final_kernel(const float* __restrict__ actL, const float* __restrict__ actR,
                             float* __restrict__ dout, int Nn) {
  int tid = blockIdx.x * blockDim.x + threadIdx.x;
  if (tid >= Nn * 512) return;
  const int n = tid / 512, cch = tid % 512;
  dout[tid] = (cch < CHN) ? actL[n * CHN + cch] : actR[n * CHN + (cch - CHN)];
}

// ------------------------- host orchestration ------------------------------
static inline int gridFor(long n, int b) { return (int)((n + b - 1) / b); }

extern "C" void kernel_launch(void* const* d_in, const int* in_sizes, int n_in,
                              void* d_out, int out_size, void* d_ws, size_t ws_size,
                              hipStream_t stream) {
  (void)in_sizes; (void)n_in; (void)out_size; (void)ws_size;
  const float* left_x    = (const float*)d_in[0];
  const float* right_x   = (const float*)d_in[1];
  const float* src_state = (const float*)d_in[2];
  const float* tgt_state = (const float*)d_in[3];
  const int*   tokens    = (const int*)d_in[4];
  const int*   left_ei   = (const int*)d_in[5];
  const int*   right_ei  = (const int*)d_in[6];
  const float* embed     = (const float*)d_in[7];
  const float* W_ih_f    = (const float*)d_in[8];
  const float* W_hh_f    = (const float*)d_in[9];
  const float* b_f       = (const float*)d_in[10];
  const float* W_ih_b    = (const float*)d_in[11];
  /* d_in[12] = W_hh_b: unused (backward dir is a single step from h=c=0) */
  const float* b_b       = (const float*)d_in[13];
  const float* g1_l      = (const float*)d_in[14];
  const float* g1_r      = (const float*)d_in[15];
  const float* g1_e      = (const float*)d_in[16];
  const float* g1_att    = (const float*)d_in[17];
  const float* g1_b      = (const float*)d_in[18];
  const float* g_l       = (const float*)d_in[19];
  const float* g_r       = (const float*)d_in[20];
  const float* g_e       = (const float*)d_in[21];
  const float* g_att     = (const float*)d_in[22];
  const float* g_b       = (const float*)d_in[23];

  char* ws = (char*)d_ws;
  size_t off = 0;
  auto take = [&](size_t bytes) { size_t o = off; off = (off + bytes + 255) & ~(size_t)255; return o; };

  float* regex  = (float*)(ws + take((size_t)EE * EDIM * 4));
  size_t o_seg0 = take((size_t)NN * ST * 4);  float* in_tr  = (float*)(ws + o_seg0);
  float* out_tr = (float*)(ws + take((size_t)NN * ST * 4));
  float* cnt_s  = (float*)(ws + take((size_t)NN * 4));
  float* cnt_d  = (float*)(ws + take((size_t)NN * 4));
  size_t segspan = off - o_seg0;
  float* lx     = (float*)(ws + take((size_t)NN * IN1 * 4));
  float* rx     = (float*)(ws + take((size_t)NN * IN1 * 4));
  float* actL   = (float*)(ws + take((size_t)NN * CHN * 4));
  float* actR   = (float*)(ws + take((size_t)NN * CHN * 4));
  float* xlb    = (float*)(ws + take((size_t)NN * CHN * 4));
  float* xrb    = (float*)(ws + take((size_t)NN * CHN * 4));
  float* outacc = (float*)(ws + take((size_t)NN * CHN * 4));
  float* eeb    = (float*)(ws + take((size_t)EE * CHN * 4));
  float* logit  = (float*)(ws + take((size_t)EE * 8 * 4));
  float* exb    = (float*)(ws + take((size_t)EE * 8 * 4));
  float* mx     = (float*)(ws + take((size_t)NN * 8 * 4));
  float* den    = (float*)(ws + take((size_t)NN * 8 * 4));
  unsigned short* wBihF = (unsigned short*)(ws + take((size_t)32 * 256 * 2));
  unsigned short* wBhhF = (unsigned short*)(ws + take((size_t)64 * 256 * 2));
  unsigned short* wBihB = (unsigned short*)(ws + take((size_t)32 * 256 * 2));
  unsigned short* wG1L  = (unsigned short*)(ws + take((size_t)IN1P * 256 * 2));
  unsigned short* wG1R  = (unsigned short*)(ws + take((size_t)IN1P * 256 * 2));
  unsigned short* wGL   = (unsigned short*)(ws + take((size_t)4 * 256 * 256 * 2));
  unsigned short* wGR   = (unsigned short*)(ws + take((size_t)4 * 256 * 256 * 2));
  unsigned short* wG1E  = (unsigned short*)(ws + take((size_t)128 * 256 * 2));
  unsigned short* wGE   = (unsigned short*)(ws + take((size_t)4 * 128 * 256 * 2));

  auto packB = [&](const float* src, unsigned short* dst, int K, int Nc, int Kpad, int trans) {
    pack_b_kernel<<<gridFor((long)Kpad * Nc, 256), 256, 0, stream>>>(src, dst, K, Nc, Kpad, trans);
  };
  packB(W_ih_f, wBihF, 32, 256, 32, 1);        // [4H,DEMB] -> transpose
  packB(W_hh_f, wBhhF, 64, 256, 64, 1);
  packB(W_ih_b, wBihB, 32, 256, 32, 1);
  packB(g1_l, wG1L, IN1, 256, IN1P, 0);        // already [K,N]
  packB(g1_r, wG1R, IN1, 256, IN1P, 0);
  for (int i = 0; i < 4; ++i) {
    packB(g_l + (size_t)i * 65536, wGL + (size_t)i * 65536, 256, 256, 256, 0);
    packB(g_r + (size_t)i * 65536, wGR + (size_t)i * 65536, 256, 256, 256, 0);
    packB(g_e + (size_t)i * 32768, wGE + (size_t)i * 32768, 128, 256, 128, 0);
  }
  packB(g1_e, wG1E, 128, 256, 128, 0);

  // 1) edge LSTM -> regex[E,128]
  lstm_kernel<<<EE / 16, 128, 0, stream>>>(tokens, embed, wBihF, wBhhF, b_f, wBihB, b_b, regex, EE);

  // 2) segment means over left_ei -> in_tr/out_tr; then lx/rx
  hipMemsetAsync(ws + o_seg0, 0, segspan, stream);
  seg_scatter_kernel<<<gridFor(2L * EE * ST, 256), 256, 0, stream>>>(
      src_state, tgt_state, regex, left_ei, in_tr, out_tr, cnt_s, cnt_d, EE);
  seg_div_kernel<<<gridFor((long)NN * ST, 256), 256, 0, stream>>>(in_tr, out_tr, cnt_s, cnt_d, NN);
  concat_kernel<<<gridFor((long)NN * IN1, 256), 256, 0, stream>>>(left_x, right_x, in_tr, out_tr, lx, rx, NN);

  // 3) 5 GAT layers, left & right interleaved (shared per-layer edge GEMM)
  const unsigned short* WlP[5] = {wG1L, wGL, wGL + 65536, wGL + 2 * 65536, wGL + 3 * 65536};
  const unsigned short* WrP[5] = {wG1R, wGR, wGR + 65536, wGR + 2 * 65536, wGR + 3 * 65536};
  const unsigned short* WeP[5] = {wG1E, wGE, wGE + 32768, wGE + 2 * 32768, wGE + 3 * 32768};
  const float* attP[5] = {g1_att, g_att, g_att + 256, g_att + 512, g_att + 768};
  const float* bP[5]   = {g1_b,   g_b,   g_b + 256,   g_b + 512,   g_b + 768};

  for (int l = 0; l < 5; ++l) {
    gemm_kernel<<<dim3(EE / 16, 4), 128, 0, stream>>>(regex, WeP[l], nullptr, eeb, EE, 128, 128, 256);
    const int K = (l == 0) ? IN1 : 256, Kp = (l == 0) ? IN1P : 256;
    for (int side = 0; side < 2; ++side) {
      const float* Ain = (l == 0) ? (side ? rx : lx) : (side ? actR : actL);
      const int* ei = side ? right_ei : left_ei;
      float* actOut = side ? actR : actL;
      gemm_kernel<<<dim3(NN / 16, 4), 128, 0, stream>>>(Ain, WlP[l], nullptr, xlb, NN, K, Kp, 256);
      gemm_kernel<<<dim3(NN / 16, 4), 128, 0, stream>>>(Ain, WrP[l], nullptr, xrb, NN, K, Kp, 256);
      att_init_kernel<<<gridFor((long)NN * CHN, 256), 256, 0, stream>>>(mx, den, outacc, bP[l], NN);
      logits_kernel<<<gridFor((long)EE * 8, 256), 256, 0, stream>>>(xlb, xrb, eeb, attP[l], ei, logit, mx, EE);
      expden_kernel<<<gridFor((long)EE * 8, 256), 256, 0, stream>>>(logit, mx, ei, exb, den, EE);
      attout_kernel<<<gridFor((long)EE * CHN, 256), 256, 0, stream>>>(xlb, exb, den, ei, outacc, EE);
      elu_kernel<<<gridFor((long)NN * CHN, 256), 256, 0, stream>>>(outacc, actOut, NN * CHN);
    }
  }

  // 4) concat [left, right] -> d_out [N, 512]
  final_kernel<<<gridFor((long)NN * 512, 256), 256, 0, stream>>>(actL, actR, (float*)d_out, NN);
}
